// BFConv2d_4698694222254
// MI455X (gfx1250) — matrix operand order repaired and checked by reading, compiled, not run on hardware
//
#include <hip/hip_runtime.h>

// ---------------------------------------------------------------------------
// BFConv2d (block-floating-point conv) for gfx1250 / MI455X
//   pass 1: quantize x  -> bf16 (exact container for BFP8), NCHW layout
//   pass 2: quantize w  -> bf16, repacked tap-major [kh*3+kw][o][i]
//   pass 3: implicit-GEMM 3x3 conv with v_wmma_f32_16x16x32_bf16
//           8 output rows per workgroup to amortize weight reads (L2 traffic)
// ---------------------------------------------------------------------------

typedef __attribute__((ext_vector_type(16))) __bf16        v16bf;
typedef __attribute__((ext_vector_type(8)))  float         v8f;
typedef __attribute__((ext_vector_type(4)))  unsigned int  v4u;

#define NB    64
#define CI    128
#define CO    128
#define HW    56
#define TAPS  9
#define ROWS  8                  // output rows per workgroup (56 % 8 == 0)
#define IROWS (ROWS + 2)         // input rows staged in LDS
#define LDSC  136                // padded c-stride (272B lane stride -> spreads banks)

union FragBF {
    v16bf v;
    v4u   u[2];
};

// exact f32 -> bf16 bit truncation (values here have <= 8 significant bits)
__device__ __forceinline__ unsigned short f32_to_bf16_bits(float f) {
    union { float f; unsigned int u; } cv;
    cv.f = f;
    return (unsigned short)(cv.u >> 16);
}

// BFP8 quantize one element given its 16-element-group shared abs-max.
// e = floor(log2(m)); scale = 2^(e-7); q = clamp(rne(v/scale), -127, 127)*scale
__device__ __forceinline__ float bfp_quant_elem(float v, float m) {
    if (!(m > 0.0f)) return 0.0f;
    union { float f; unsigned int u; } mb;
    mb.f = m;
    int e = (int)((mb.u >> 23) & 0xFF) - 127;       // floor(log2(m)) for normals
    union { float f; unsigned int u; } sc, isc;
    sc.u  = (unsigned int)(e - 7 + 127) << 23;      // scale   = 2^(e-7)
    isc.u = (unsigned int)(7 - e + 127) << 23;      // 1/scale = 2^(7-e), exact
    float q = rintf(v * isc.f);                     // round-to-nearest-even
    q = fminf(fmaxf(q, -127.0f), 127.0f);
    return q * sc.f;
}

// group-of-16 abs-max reduction (flat groups align with 16-lane halves of wave32)
__device__ __forceinline__ float group16_absmax(float v) {
    float a = fabsf(v);
    a = fmaxf(a, __shfl_xor(a, 1, 16));
    a = fmaxf(a, __shfl_xor(a, 2, 16));
    a = fmaxf(a, __shfl_xor(a, 4, 16));
    a = fmaxf(a, __shfl_xor(a, 8, 16));
    return a;
}

__global__ __launch_bounds__(256) void bfp_quant_x(const float* __restrict__ x,
                                                   unsigned short* __restrict__ xq,
                                                   int n) {
    int idx = blockIdx.x * blockDim.x + threadIdx.x;
    if (idx >= n) return;
    float v = x[idx];
    float m = group16_absmax(v);
    xq[idx] = f32_to_bf16_bits(bfp_quant_elem(v, m));
}

// quantize in flat OIHW group order, scatter to tap-major [kh*3+kw][o][i]
__global__ __launch_bounds__(256) void bfp_quant_w_repack(const float* __restrict__ w,
                                                          unsigned short* __restrict__ wq,
                                                          int n) {
    int idx = blockIdx.x * blockDim.x + threadIdx.x;
    if (idx >= n) return;
    float v = w[idx];
    float m = group16_absmax(v);
    unsigned short bits = f32_to_bf16_bits(bfp_quant_elem(v, m));
    int kw = idx % 3;
    int kh = (idx / 3) % 3;
    int i  = (idx / 9) % CI;
    int o  = idx / (9 * CI);
    wq[((kh * 3 + kw) * CO + o) * CI + i] = bits;
}

// ---------------------------------------------------------------------------
// Implicit-GEMM conv. Block = 256 threads = 8 waves.
//   blockIdx = (q-strip 0..3, p-tile 0..6, n 0..63)
//   wave w -> output channels [16w, 16w+16), 16 output columns, 8 output rows
//   reduction: 9 taps x 4 chunks of 32 input channels; A fragment reused
//   across the 8 output rows -> 288 WMMAs / wave, 8 independent acc chains
// ---------------------------------------------------------------------------
__global__ __launch_bounds__(256) void conv_wmma(const unsigned short* __restrict__ xq,
                                                 const unsigned short* __restrict__ wq,
                                                 float* __restrict__ out) {
    __shared__ unsigned short smem[IROWS * 18 * LDSC];   // 48,960 B

    const int q0  = blockIdx.x * 16;
    const int p0  = blockIdx.y * ROWS;
    const int n   = blockIdx.z;
    const int tid = threadIdx.x;

    // ---- stage input tile: rows p0-1 .. p0+ROWS, cols q0-1..q0+16, all c ----
    // iterate (row,col) fastest so global reads are contiguous per (c,row)
    for (int t = tid; t < IROWS * 18 * CI; t += 256) {
        int c   = t / (IROWS * 18);
        int rc  = t - c * (IROWS * 18);
        int row = rc / 18;
        int col = rc - row * 18;
        int h   = p0 - 1 + row;
        int wa  = q0 - 1 + col;
        unsigned short v = 0;
        if ((unsigned)h < (unsigned)HW && (unsigned)wa < (unsigned)HW)
            v = xq[((n * CI + c) * HW + h) * HW + wa];
        smem[(row * 18 + col) * LDSC + c] = v;
    }
    __syncthreads();

    const int wave    = tid >> 5;
    const int lane    = tid & 31;
    const int lane16  = lane & 15;
    const int halfsel = lane >> 4;   // 0: lanes 0-15, 1: lanes 16-31
    const int ko      = wave * 16;

    v8f acc[ROWS];
#pragma unroll
    for (int r = 0; r < ROWS; ++r) acc[r] = (v8f){};

    for (int tap = 0; tap < TAPS; ++tap) {
        const int kh = tap / 3;
        const int kw = tap - kh * 3;
        // A (weights): row M = ko+lane16; K split {0..7,16..23} / {8..15,24..31}
        const unsigned short* wbase =
            wq + ((tap * CO + ko + lane16) * CI + halfsel * 8);
        // B (inputs): column N = lane16 -> LDS col lane16+kw;
        //             K split {0..7,16..23} / {8..15,24..31}
        const unsigned short* bcol =
            smem + ((kh * 18) + lane16 + kw) * LDSC + halfsel * 8;

#pragma unroll
        for (int cc = 0; cc < 4; ++cc) {
            FragBF a;
            a.u[0] = *(const v4u*)(wbase + cc * 32);        // K 0..7 / 8..15
            a.u[1] = *(const v4u*)(wbase + cc * 32 + 16);   // K 16..23 / 24..31

            const unsigned short* bb = bcol + cc * 32;
#pragma unroll
            for (int r = 0; r < ROWS; ++r) {
                FragBF b;
                b.u[0] = *(const v4u*)(bb + r * (18 * LDSC));
                b.u[1] = *(const v4u*)(bb + r * (18 * LDSC) + 16);
                acc[r] = __builtin_amdgcn_wmma_f32_16x16x32_bf16(
                    false, a.v, false, b.v, (short)0, acc[r], false, false);
            }
        }
    }

    // ---- store: VGPR j -> M = j + halfsel*8, N = lane16 ----
    const int q = q0 + lane16;
    if (q < HW) {
        const int kbase = ko + halfsel * 8;
#pragma unroll
        for (int j = 0; j < 8; ++j) {
            size_t base = ((size_t)n * CO + (kbase + j)) * HW;
#pragma unroll
            for (int r = 0; r < ROWS; ++r) {
                out[(base + (p0 + r)) * HW + q] = acc[r][j];
            }
        }
    }
}

extern "C" void kernel_launch(void* const* d_in, const int* in_sizes, int n_in,
                              void* d_out, int out_size, void* d_ws, size_t ws_size,
                              hipStream_t stream) {
    (void)n_in; (void)out_size; (void)ws_size;
    const float* x = (const float*)d_in[0];
    const float* w = (const float*)d_in[1];
    float* out = (float*)d_out;

    const int nx = in_sizes[0];   // 64*128*56*56 = 25,690,112
    const int nw = in_sizes[1];   // 128*128*3*3  = 147,456

    unsigned short* xq = (unsigned short*)d_ws;
    size_t xbytes = (((size_t)nx * sizeof(unsigned short)) + 255) & ~(size_t)255;
    unsigned short* wq = (unsigned short*)((char*)d_ws + xbytes);

    bfp_quant_x<<<(nx + 255) / 256, 256, 0, stream>>>(x, xq, nx);
    bfp_quant_w_repack<<<(nw + 255) / 256, 256, 0, stream>>>(w, wq, nw);

    dim3 grid((HW + 15) / 16, HW / ROWS, NB);   // (4, 7, 64)
    conv_wmma<<<grid, 256, 0, stream>>>(xq, wq, out);
}
